// Attn_Net_Gated_31842887532811
// MI455X (gfx1250) — matrix operand order repaired and checked by reading, not compiled, not run on hardware
//
#include <hip/hip_runtime.h>
#include <hip/hip_bf16.h>

#define N_ROWS 100000
#define DIM    1024
#define LDIM   128
#define NBATCH 16
#define EPSF   1e-16f
#define BROWS  64
#define LDSS   1032   // padded bf16 row stride (spreads LDS banks)
#define SROWS  256

typedef __attribute__((ext_vector_type(16))) __bf16 v16bf;
typedef __attribute__((ext_vector_type(8)))  float  v8f;

union CFrag  { v8f v;   float f[8]; };
union Frag16 { v16bf v; uint4 q[2]; };

__device__ __forceinline__ __bf16 f32_to_bf16(float x) {
  union { float f; unsigned u; } in{x};
  unsigned r = in.u + 0x7FFFu + ((in.u >> 16) & 1u);   // round-to-nearest-even
  union { unsigned short s; __bf16 b; } out{(unsigned short)(r >> 16)};
  return out.b;
}

// order-preserving float <-> uint encoding for atomicMax on floats
__device__ __forceinline__ unsigned encf(float f) {
  union { float f; unsigned u; } v{f};
  return (v.u & 0x80000000u) ? ~v.u : (v.u | 0x80000000u);
}
__device__ __forceinline__ float decf(unsigned u) {
  unsigned b = (u & 0x80000000u) ? (u & 0x7FFFFFFFu) : ~u;
  union { unsigned u; float f; } v{b};
  return v.f;
}
#define ENC_NEG_INF 0x007FFFFFu

// ---------------------------------------------------------------- init
__global__ void init_kernel(float* __restrict__ out_acc,
                            unsigned* __restrict__ seg_max,
                            float* __restrict__ denom) {
  int i = blockIdx.x * blockDim.x + threadIdx.x;
  if (i < NBATCH * DIM) out_acc[i] = 0.f;
  if (i < NBATCH) { seg_max[i] = ENC_NEG_INF; denom[i] = 0.f; }
}

// ------------------------------------------------------ weights -> bf16
__global__ void cvt_weights(const float* __restrict__ Wa,
                            const float* __restrict__ Wb,
                            __bf16* __restrict__ wa_bf,
                            __bf16* __restrict__ wb_bf) {
  int i = blockIdx.x * blockDim.x + threadIdx.x;   // 0 .. 131071
  wa_bf[i] = f32_to_bf16(Wa[i]);
  wb_bf[i] = f32_to_bf16(Wb[i]);
}

// ------------------------------------- fused WMMA GEMM + gate + score
__global__ __launch_bounds__(256)
void gemm_score(const float*  __restrict__ feature,
                const __bf16* __restrict__ wa_bf,
                const __bf16* __restrict__ wb_bf,
                const float*  __restrict__ ba,
                const float*  __restrict__ bb,
                const float*  __restrict__ Wc,
                const float*  __restrict__ bc,
                float*        __restrict__ score_out) {
  __shared__ __bf16 ldsA[BROWS * LDSS];
  __shared__ float  sScore[BROWS];

  const int tid  = threadIdx.x;
  const int row0 = blockIdx.x * BROWS;

  // ---- stage 64 rows x 1024 cols of feature into LDS as bf16 ----
  for (int it = 0; it < 64; ++it) {
    int idx = tid + 256 * it;          // 0 .. 16383 float4 tiles
    int r   = idx >> 8;                // 0 .. 63
    int c4  = (idx & 255) << 2;        // col (multiple of 4)
    int grow = row0 + r;
    float4 f;
    if (grow < N_ROWS) f = *(const float4*)(feature + (size_t)grow * DIM + c4);
    else               f = make_float4(0.f, 0.f, 0.f, 0.f);
    union { __bf16 b[4]; uint2 u; } pk;
    pk.b[0] = f32_to_bf16(f.x); pk.b[1] = f32_to_bf16(f.y);
    pk.b[2] = f32_to_bf16(f.z); pk.b[3] = f32_to_bf16(f.w);
    *(uint2*)&ldsA[r * LDSS + c4] = pk.u;
  }
  if (tid < BROWS) sScore[tid] = 0.f;
  __syncthreads();

  const int wave = tid >> 5;           // 0..7  -> L-tile
  const int lane = tid & 31;
  const int nidx = lane & 15;
  const int half = lane >> 4;
  const int l0   = wave * 16;

  CFrag accA[4], accB[4];
#pragma unroll
  for (int mt = 0; mt < 4; ++mt) {
    v8f z = {0.f,0.f,0.f,0.f,0.f,0.f,0.f,0.f};
    accA[mt].v = z; accB[mt].v = z;
  }

  // ---- K loop: 1024 in steps of 32 (bf16 WMMA 16x16x32) ----
  for (int k = 0; k < DIM; k += 32) {
    // B frags: W row l = l0+nidx, 16 consecutive K at half*16 (contiguous k order, same as A frag)
    const __bf16* pwa = wa_bf + (size_t)(l0 + nidx) * DIM + k + half * 16;
    const __bf16* pwb = wb_bf + (size_t)(l0 + nidx) * DIM + k + half * 16;
    Frag16 wfa, wfb;
    wfa.q[0] = *(const uint4*)pwa;      wfa.q[1] = *(const uint4*)(pwa + 8);
    wfb.q[0] = *(const uint4*)pwb;      wfb.q[1] = *(const uint4*)(pwb + 8);
#pragma unroll
    for (int mt = 0; mt < 4; ++mt) {
      // A frag: row m = mt*16+nidx; elems 0-15 K=half*16+j
      const __bf16* pa = &ldsA[(mt * 16 + nidx) * LDSS + k + half * 16];
      Frag16 af;
      af.q[0] = *(const uint4*)pa;
      af.q[1] = *(const uint4*)(pa + 8);
      accA[mt].v = __builtin_amdgcn_wmma_f32_16x16x32_bf16(
          false, af.v, false, wfa.v, (short)0, accA[mt].v, false, false);
      accB[mt].v = __builtin_amdgcn_wmma_f32_16x16x32_bf16(
          false, af.v, false, wfb.v, (short)0, accB[mt].v, false, false);
    }
  }

  // ---- epilogue: bias + sigmoid*tanh, dot with Wc, row-reduce ----
  const int   l    = l0 + nidx;
  const float ba_l = ba[l], bb_l = bb[l], wc_l = Wc[l];
#pragma unroll
  for (int mt = 0; mt < 4; ++mt) {
#pragma unroll
    for (int r = 0; r < 8; ++r) {
      // C/D layout: VGPR r, lanes 0-15 -> M=r, lanes 16-31 -> M=8+r; N=lane&15
      float av = accA[mt].f[r] + ba_l;
      float bv = accB[mt].f[r] + bb_l;
      float g  = (1.0f / (1.0f + __expf(-av))) * tanhf(bv);
      float p  = g * wc_l;
      p += __shfl_xor(p, 1);
      p += __shfl_xor(p, 2);
      p += __shfl_xor(p, 4);
      p += __shfl_xor(p, 8);       // row-sum within each 16-lane half
      if (nidx == 0) atomicAdd(&sScore[mt * 16 + r + half * 8], p);
    }
  }
  __syncthreads();
  if (tid < BROWS) {
    int grow = row0 + tid;
    if (grow < N_ROWS) score_out[grow] = sScore[tid] + bc[0];
  }
}

// ---------------------------------------------------------- segment max
__global__ void segmax_kernel(const float* __restrict__ score,
                              const long long* __restrict__ batch,
                              unsigned* __restrict__ seg_max, int n) {
  __shared__ unsigned lmax[NBATCH];
  if (threadIdx.x < NBATCH) lmax[threadIdx.x] = ENC_NEG_INF;
  __syncthreads();
  for (int i = blockIdx.x * blockDim.x + threadIdx.x; i < n;
       i += gridDim.x * blockDim.x) {
    atomicMax(&lmax[(int)batch[i]], encf(score[i]));
  }
  __syncthreads();
  if (threadIdx.x < NBATCH) atomicMax(&seg_max[threadIdx.x], lmax[threadIdx.x]);
}

// --------------------------------------------------- segment exp-sum
__global__ void denom_kernel(const float* __restrict__ score,
                             const long long* __restrict__ batch,
                             const unsigned* __restrict__ seg_max,
                             float* __restrict__ denom, int n) {
  __shared__ float lsum[NBATCH];
  if (threadIdx.x < NBATCH) lsum[threadIdx.x] = 0.f;
  __syncthreads();
  for (int i = blockIdx.x * blockDim.x + threadIdx.x; i < n;
       i += gridDim.x * blockDim.x) {
    int b = (int)batch[i];
    atomicAdd(&lsum[b], __expf(score[i] - decf(seg_max[b])));
  }
  __syncthreads();
  if (threadIdx.x < NBATCH) atomicAdd(&denom[threadIdx.x], lsum[threadIdx.x]);
}

// ---- softmax + weighted segment-sum + feature copy (rows sorted) ----
__global__ __launch_bounds__(256)
void segsum_kernel(const float* __restrict__ feature,
                   const long long* __restrict__ batch,
                   const float* __restrict__ score,
                   const unsigned* __restrict__ seg_max,
                   const float* __restrict__ denom,
                   float* __restrict__ out_acc,
                   float* __restrict__ ssm_out,
                   float* __restrict__ feat_out, int n) {
  const int tid = threadIdx.x;
  const int c   = tid * 4;              // 4 columns per thread
  int r0 = blockIdx.x * SROWS;
  if (r0 >= n) return;
  int r1 = r0 + SROWS; if (r1 > n) r1 = n;

  float4 a = make_float4(0.f, 0.f, 0.f, 0.f);
  int cur = (int)batch[r0];
  for (int r = r0; r < r1; ++r) {
    int b = (int)batch[r];              // uniform across block
    if (b != cur) {
      atomicAdd(&out_acc[cur * DIM + c + 0], a.x);
      atomicAdd(&out_acc[cur * DIM + c + 1], a.y);
      atomicAdd(&out_acc[cur * DIM + c + 2], a.z);
      atomicAdd(&out_acc[cur * DIM + c + 3], a.w);
      a = make_float4(0.f, 0.f, 0.f, 0.f);
      cur = b;
    }
    float s   = score[r];
    float ssm = __expf(s - decf(seg_max[b])) / (denom[b] + EPSF);
    if (tid == 0) ssm_out[r] = ssm;
    float4 f = *(const float4*)(feature + (size_t)r * DIM + c);
    *(float4*)(feat_out + (size_t)r * DIM + c) = f;  // feature copy output
    a.x += ssm * f.x; a.y += ssm * f.y; a.z += ssm * f.z; a.w += ssm * f.w;
  }
  atomicAdd(&out_acc[cur * DIM + c + 0], a.x);
  atomicAdd(&out_acc[cur * DIM + c + 1], a.y);
  atomicAdd(&out_acc[cur * DIM + c + 2], a.z);
  atomicAdd(&out_acc[cur * DIM + c + 3], a.w);
}

// ------------------------------------------------------------- launch
extern "C" void kernel_launch(void* const* d_in, const int* in_sizes, int n_in,
                              void* d_out, int out_size, void* d_ws, size_t ws_size,
                              hipStream_t stream) {
  const float*     feature = (const float*)d_in[0];
  const long long* batch   = (const long long*)d_in[1];
  const float*     Wa      = (const float*)d_in[2];
  const float*     ba      = (const float*)d_in[3];
  const float*     Wb      = (const float*)d_in[4];
  const float*     bb      = (const float*)d_in[5];
  const float*     Wc      = (const float*)d_in[6];
  const float*     bc      = (const float*)d_in[7];

  float* out    = (float*)d_out;                 // (16, 1024)
  float* score  = out + NBATCH * DIM;            // (N,)
  float* ssm    = score + N_ROWS;                // (N, 1)
  float* featc  = ssm + N_ROWS;                  // (N, 1024)

  char* ws = (char*)d_ws;
  __bf16*   wa_bf   = (__bf16*)ws;                              // 256 KB
  __bf16*   wb_bf   = (__bf16*)(ws + (size_t)LDIM * DIM * 2);   // 256 KB
  unsigned* seg_max = (unsigned*)(ws + (size_t)2 * LDIM * DIM * 2);
  float*    denom   = (float*)(seg_max + NBATCH);

  init_kernel<<<64, 256, 0, stream>>>(out, seg_max, denom);
  cvt_weights<<<(LDIM * DIM) / 256, 256, 0, stream>>>(Wa, Wb, wa_bf, wb_bf);
  gemm_score<<<(N_ROWS + BROWS - 1) / BROWS, 256, 0, stream>>>(
      feature, wa_bf, wb_bf, ba, bb, Wc, bc, score);
  segmax_kernel<<<256, 256, 0, stream>>>(score, batch, seg_max, N_ROWS);
  denom_kernel<<<256, 256, 0, stream>>>(score, batch, seg_max, denom, N_ROWS);
  segsum_kernel<<<(N_ROWS + SROWS - 1) / SROWS, 256, 0, stream>>>(
      feature, batch, score, seg_max, denom, out, ssm, featc, N_ROWS);
}
